// GaussianLSS_72086731096597
// MI455X (gfx1250) — matrix operand order repaired and checked by reading, not compiled, not run on hardware
//
#include <hip/hip_runtime.h>
#include <hip/hip_bf16.h>
#include <math.h>

typedef __attribute__((ext_vector_type(16))) _Float16 v16h;
typedef __attribute__((ext_vector_type(8)))  _Float16 v8h;
typedef __attribute__((ext_vector_type(8)))  float    v8f;

// Problem constants: B=1, N=6, C=256, H=32, W=88 (HW=2816), D=44, conv3 out = 130 ch
#define NCAM 6
#define CCH  256
#define HH   32
#define WW   88
#define HWP  2816
#define DD   44
#define GEOM_ELEMS ((size_t)NCAM * DD * HWP)   // points (x3 floats)

// ---------------- small 3x3 helpers ----------------
__device__ inline void inv3(const float* a, float* o) {
    float a00=a[0],a01=a[1],a02=a[2];
    float a10=a[3],a11=a[4],a12=a[5];
    float a20=a[6],a21=a[7],a22=a[8];
    float det = a00*(a11*a22-a12*a21) - a01*(a10*a22-a12*a20) + a02*(a10*a21-a11*a20);
    float id = 1.0f/det;
    o[0]=(a11*a22-a12*a21)*id; o[1]=(a02*a21-a01*a22)*id; o[2]=(a01*a12-a02*a11)*id;
    o[3]=(a12*a20-a10*a22)*id; o[4]=(a00*a22-a02*a20)*id; o[5]=(a02*a10-a00*a12)*id;
    o[6]=(a10*a21-a11*a20)*id; o[7]=(a01*a20-a00*a21)*id; o[8]=(a00*a11-a01*a10)*id;
}

// cam[n*18 + 0..8] = inv(prot), cam[n*18 + 9..17] = rot @ inv(K)
__global__ void cam_setup(const float* __restrict__ rot, const float* __restrict__ K,
                          const float* __restrict__ prot, float* __restrict__ cam) {
    int n = threadIdx.x;
    if (n >= NCAM) return;
    float iP[9], iK[9];
    inv3(prot + n*9, iP);
    inv3(K    + n*9, iK);
    const float* R = rot + n*9;
    float M[9];
    #pragma unroll
    for (int i = 0; i < 3; ++i)
        #pragma unroll
        for (int j = 0; j < 3; ++j)
            M[i*3+j] = R[i*3+0]*iK[0+j] + R[i*3+1]*iK[3+j] + R[i*3+2]*iK[6+j];
    #pragma unroll
    for (int i = 0; i < 9; ++i) { cam[n*18+i] = iP[i]; cam[n*18+9+i] = M[i]; }
}

// ---------------- geometry ----------------
__global__ void geom_kernel(const float* __restrict__ cam,
                            const float* __restrict__ ptrans,
                            const float* __restrict__ trans,
                            float* __restrict__ geom) {
    size_t idx = (size_t)blockIdx.x * blockDim.x + threadIdx.x;
    if (idx >= GEOM_ELEMS) return;
    int n   = (int)(idx / ((size_t)DD * HWP));
    int rem = (int)(idx % ((size_t)DD * HWP));
    int d = rem / HWP;
    int p = rem % HWP;
    int h = p / WW, w = p % WW;
    const float* iP = cam + n*18;
    const float* M  = cam + n*18 + 9;
    float fx = (float)w * (703.0f/87.0f);   // linspace(0, ORW-1, W)
    float fy = (float)h * (255.0f/31.0f);   // linspace(0, ORH-1, H)
    float fd = (float)(d + 1);              // DV = 1..44
    float px = fx - ptrans[n*3+0];
    float py = fy - ptrans[n*3+1];
    float pz = fd - ptrans[n*3+2];
    float qx = iP[0]*px + iP[1]*py + iP[2]*pz;
    float qy = iP[3]*px + iP[4]*py + iP[5]*pz;
    float qz = iP[6]*px + iP[7]*py + iP[8]*pz;
    float rx = qx*qz, ry = qy*qz, rz = qz;
    geom[idx*3+0] = M[0]*rx + M[1]*ry + M[2]*rz + trans[n*3+0];
    geom[idx*3+1] = M[3]*rx + M[4]*ry + M[5]*rz + trans[n*3+1];
    geom[idx*3+2] = M[6]*rx + M[7]*ry + M[8]*rz + trans[n*3+2];
}

// ---------------- conversions ----------------
__global__ void cvt_f32_f16(const float* __restrict__ in, _Float16* __restrict__ o, int nElem) {
    int i = blockIdx.x * 256 + threadIdx.x;
    if (i < nElem) o[i] = (_Float16)in[i];
}

// w (OC=256, IC=256, 3, 3) OIHW  ->  wk[r][oc][ic], r = kh*3+kw
__global__ void cvt_w3x3(const float* __restrict__ w, _Float16* __restrict__ o) {
    int i = blockIdx.x * 256 + threadIdx.x;   // over 9*256*256 = 589824
    int r   = i >> 16;
    int rem = i & 65535;
    int oc  = rem >> 8;
    int ic  = rem & 255;
    o[i] = (_Float16)w[(size_t)(oc * 256 + ic) * 9 + r];
}

// w3 (130, 256) -> padded [144][256] f16 (rows >=130 zero)
__global__ void cvt_w1x1(const float* __restrict__ w, _Float16* __restrict__ o) {
    int i = blockIdx.x * 256 + threadIdx.x;   // over 144*256 = 36864
    int oc = i >> 8, ic = i & 255;
    o[i] = (_Float16)((oc < 130) ? w[oc * 256 + ic] : 0.0f);
}

// ---------------- conv 3x3 + BN + ReLU (implicit GEMM, WMMA f16) ----------------
// Grid: (88 pixel-tiles of 32, 4 oc-tiles of 64, 6 images). Block: 256 thr = 8 waves.
// Double-buffered LDS software pipeline; B tile stored TRANSPOSED [px][ic] so
// each lane's WMMA B fragment is two 16-byte runs at ic 8*hs and 16+8*hs (2x ds_load_b128).
#define KSTEPS 72   // 9 taps * (256/32) ic-chunks

__global__ __launch_bounds__(256)
void conv3x3_bn_relu(const _Float16* __restrict__ xin,   // [6][256][2816] f16
                     const _Float16* __restrict__ wk,    // [9][256][256]  f16 (r, oc, ic)
                     const float* __restrict__ bias,
                     const float* __restrict__ gam,  const float* __restrict__ bet,
                     const float* __restrict__ mean, const float* __restrict__ var,
                     _Float16* __restrict__ yout)        // [6][256][2816] f16
{
    __shared__ __align__(16) _Float16 At[2][64][40];   // [buf][oc][ic], stride 80B
    __shared__ __align__(16) _Float16 Bt[2][32][40];   // [buf][px][ic] (transposed!)

    const int tid  = threadIdx.x;
    const int lane = tid & 31;
    const int wave = tid >> 5;
    const int n    = blockIdx.z;
    const int oc0  = blockIdx.y * 64;
    const int p0   = blockIdx.x * 32;
    const int ocb  = (wave & 3) * 16;
    const int pxb  = (wave >> 2) * 16;

    const int arow  = tid >> 2;          // 0..63 : A oc row
    const int acol0 = (tid & 3) * 8;     // 8 contiguous ic each
    const int brow  = tid >> 3;          // 0..31 : B ic row (coalesced global reads)
    const int bcol0 = (tid & 7) * 4;     // 4 pixels each

    // fetch one K-step's tiles from global into registers (branchless padding)
    auto loadG = [&](int step, v8h& areg, _Float16 breg[4]) {
        const int r  = step >> 3;
        const int kc = (step & 7) * 32;
        areg = *(const v8h*)(wk + ((size_t)r << 16)
                                + (size_t)(oc0 + arow) * 256 + kc + acol0);
        const int dh = r / 3 - 1;
        const int dw = r % 3 - 1;
        const _Float16* xrow = xin + (size_t)(n * 256 + kc + brow) * HWP;
        #pragma unroll
        for (int i = 0; i < 4; ++i) {
            const int p  = p0 + bcol0 + i;
            const int h  = p / WW;
            const int w  = p % WW;
            const int ih = h + dh;
            const int iw = w + dw;
            const bool ok = ((unsigned)ih < (unsigned)HH) & ((unsigned)iw < (unsigned)WW);
            const int ihc = ok ? ih : 0;
            const int iwc = ok ? iw : 0;
            _Float16 bv = xrow[ihc * WW + iwc];
            breg[i] = ok ? bv : (_Float16)0.0f;
        }
    };
    auto storeL = [&](int buf, const v8h& areg, const _Float16 breg[4]) {
        *(v8h*)&At[buf][arow][acol0] = areg;                 // one ds_store_b128
        #pragma unroll
        for (int i = 0; i < 4; ++i)
            Bt[buf][bcol0 + i][brow] = breg[i];              // transposed scatter
    };

    v8f acc = {};
    const int mrow = lane & 15;
    const int hs   = lane >> 4;

    // prologue
    v8h aP; _Float16 bP[4];
    loadG(0, aP, bP);
    storeL(0, aP, bP);
    __syncthreads();

    for (int step = 0; step < KSTEPS; ++step) {
        const int cur = step & 1;
        // issue next tile's global loads (overlaps with LDS reads + WMMA)
        v8h aN; _Float16 bN[4];
        if (step + 1 < KSTEPS) loadG(step + 1, aN, bN);

        // fragments: contiguous 16B LDS chunks per the CDNA5 16-bit layouts
        v8h alo = *(const v8h*)&At[cur][ocb + mrow][hs * 8];
        v8h ahi = *(const v8h*)&At[cur][ocb + mrow][16 + hs * 8];
        v8h blo = *(const v8h*)&Bt[cur][pxb + mrow][hs * 8];
        v8h bhi = *(const v8h*)&Bt[cur][pxb + mrow][16 + hs * 8];
        v16h a = __builtin_shufflevector(alo, ahi, 0,1,2,3,4,5,6,7,8,9,10,11,12,13,14,15);
        v16h b = __builtin_shufflevector(blo, bhi, 0,1,2,3,4,5,6,7,8,9,10,11,12,13,14,15);

        if (step + 1 < KSTEPS) storeL(cur ^ 1, aN, bN);

        acc = __builtin_amdgcn_wmma_f32_16x16x32_f16(false, a, false, b,
                                                     (short)0, acc, false, false);
        __syncthreads();
    }

    // Epilogue: bias -> BN -> ReLU -> f16
    const int p = p0 + pxb + mrow;
    #pragma unroll
    for (int v = 0; v < 8; ++v) {
        const int oc = oc0 + ocb + hs * 8 + v;
        float s   = gam[oc] * rsqrtf(var[oc] + 1e-3f);
        float val = (acc[v] + bias[oc] - mean[oc]) * s + bet[oc];
        val = val > 0.0f ? val : 0.0f;
        yout[(size_t)(n * 256 + oc) * HWP + p] = (_Float16)val;
    }
}

// ---------------- conv 1x1 (256 -> 130) via WMMA, one wave per 16x16 tile -----------
__global__ __launch_bounds__(32)
void conv1x1(const _Float16* __restrict__ xin,  // [6][256][2816] f16
             const _Float16* __restrict__ w3,   // [144][256] f16 (padded)
             const float* __restrict__ b3,
             float* __restrict__ out)           // [6][130][2816] f32
{
    const int lane = threadIdx.x & 31;
    const int n   = blockIdx.z;
    const int oc0 = blockIdx.y * 16;   // 0..128 (9 tiles)
    const int p0  = blockIdx.x * 16;   // 176 tiles
    const int m   = lane & 15;
    const int hs  = lane >> 4;

    v8f acc = {};
    for (int kc = 0; kc < 256; kc += 32) {
        // A: two contiguous 16B vector loads from the padded weight matrix
        const _Float16* wr = w3 + (size_t)(oc0 + m) * 256 + kc;
        v8h alo = *(const v8h*)(wr + hs * 16);
        v8h ahi = *(const v8h*)(wr + hs * 16 + 8);
        v16h a = __builtin_shufflevector(alo, ahi, 0,1,2,3,4,5,6,7,8,9,10,11,12,13,14,15);
        v16h b;
        #pragma unroll
        for (int v = 0; v < 8; ++v) {
            int kk = kc + 2 * v + hs * 16;
            b[2*v]   = xin[(size_t)(n * 256 + kk) * HWP + p0 + m];
            b[2*v+1] = xin[(size_t)(n * 256 + kk + 1) * HWP + p0 + m];
        }
        acc = __builtin_amdgcn_wmma_f32_16x16x32_f16(false, a, false, b,
                                                     (short)0, acc, false, false);
    }
    const int p = p0 + m;
    #pragma unroll
    for (int v = 0; v < 8; ++v) {
        const int oc = oc0 + hs * 8 + v;
        if (oc < 130)
            out[(size_t)(n * 130 + oc) * HWP + p] = acc[v] + b3[oc];
    }
}

// ---------------- Gaussian depth splat ----------------
// One block per (n, pixel): 44 exps, normalize, write 44*128 outputs (coalesced per d).
__global__ __launch_bounds__(256)
void gauss_kernel(const float* __restrict__ c3, float* __restrict__ out) {
    __shared__ float gs[DD];
    __shared__ float gsum;
    const int bid = blockIdx.x;           // 6*2816
    const int n   = bid / HWP;
    const int p   = bid % HWP;
    const int tid = threadIdx.x;

    const float mu    = c3[(size_t)(n * 130 + 0) * HWP + p];
    const float ls    = c3[(size_t)(n * 130 + 1) * HWP + p];
    const float sigma = expf(ls) + 1e-6f;

    if (tid < DD) {
        float dv = (float)(tid + 1);
        float t  = (dv - mu) / sigma;
        gs[tid]  = expf(-0.5f * t * t);
    }
    __syncthreads();
    if (tid == 0) {
        float s = 0.0f;
        #pragma unroll
        for (int d = 0; d < DD; ++d) s += gs[d];
        gsum = s + 1e-6f;
    }
    __syncthreads();

    const int oc   = tid & 127;
    const int dpar = tid >> 7;            // 0 or 1
    const float f  = c3[(size_t)(n * 130 + 2 + oc) * HWP + p];
    const float inv = 1.0f / gsum;
    for (int d = dpar; d < DD; d += 2) {
        out[((size_t)(n * DD + d) * HWP + p) * 128 + oc] = gs[d] * inv * f;
    }
}

// ---------------- launcher ----------------
extern "C" void kernel_launch(void* const* d_in, const int* in_sizes, int n_in,
                              void* d_out, int out_size, void* d_ws, size_t ws_size,
                              hipStream_t stream) {
    const float* rot    = (const float*)d_in[0];
    const float* trans  = (const float*)d_in[1];
    const float* K      = (const float*)d_in[2];
    const float* prot   = (const float*)d_in[3];
    const float* ptrans = (const float*)d_in[4];
    const float* feats  = (const float*)d_in[5];
    const float* w1 = (const float*)d_in[6];  const float* b1  = (const float*)d_in[7];
    const float* g1 = (const float*)d_in[8];  const float* be1 = (const float*)d_in[9];
    const float* m1 = (const float*)d_in[10]; const float* v1  = (const float*)d_in[11];
    const float* w2 = (const float*)d_in[12]; const float* b2  = (const float*)d_in[13];
    const float* g2 = (const float*)d_in[14]; const float* be2 = (const float*)d_in[15];
    const float* m2 = (const float*)d_in[16]; const float* v2  = (const float*)d_in[17];
    const float* w3 = (const float*)d_in[18]; const float* b3  = (const float*)d_in[19];
    float* out = (float*)d_out;

    // Workspace layout (~27 MB); every region fully written before read
    char* ws = (char*)d_ws;
    size_t off = 0;
    const size_t XHALVES = (size_t)NCAM * CCH * HWP;     // 4,325,376
    _Float16* xbuf = (_Float16*)(ws + off); off += XHALVES * 2;          // x / later y2
    _Float16* y1   = (_Float16*)(ws + off); off += XHALVES * 2;
    _Float16* wk1  = (_Float16*)(ws + off); off += (size_t)589824 * 2;
    _Float16* wk2  = (_Float16*)(ws + off); off += (size_t)589824 * 2;
    _Float16* wk3  = (_Float16*)(ws + off); off += (size_t)36864 * 2;
    float*    c3   = (float*)(ws + off);    off += (size_t)NCAM * 130 * HWP * 4;
    float*    cam  = (float*)(ws + off);    off += (size_t)NCAM * 18 * 4;

    // geometry path
    cam_setup<<<1, 32, 0, stream>>>(rot, K, prot, cam);
    geom_kernel<<<(int)((GEOM_ELEMS + 255) / 256), 256, 0, stream>>>(cam, ptrans, trans, out);

    // precision conversion / weight repack
    cvt_f32_f16<<<(int)(XHALVES / 256), 256, 0, stream>>>(feats, xbuf, (int)XHALVES);
    cvt_w3x3<<<2304, 256, 0, stream>>>(w1, wk1);
    cvt_w3x3<<<2304, 256, 0, stream>>>(w2, wk2);
    cvt_w1x1<<<144, 256, 0, stream>>>(w3, wk3);

    // conv stack (WMMA)
    dim3 cg(HWP / 32, 4, NCAM);
    conv3x3_bn_relu<<<cg, 256, 0, stream>>>(xbuf, wk1, b1, g1, be1, m1, v1, y1);
    conv3x3_bn_relu<<<cg, 256, 0, stream>>>(y1,   wk2, b2, g2, be2, m2, v2, xbuf); // y2 aliases x
    dim3 c1g(HWP / 16, 9, NCAM);
    conv1x1<<<c1g, 32, 0, stream>>>(xbuf, wk3, b3, c3);

    // Gaussian splat into second output region
    float* out2 = out + GEOM_ELEMS * 3;   // geom = points*3 floats
    gauss_kernel<<<NCAM * HWP, 256, 0, stream>>>(c3, out2);
}
